// _DCNASPPModule_71451075936611
// MI455X (gfx1250) — compile-verified, matrix operand order checked by reading, not run on hardware
//
#include <hip/hip_runtime.h>
#include <math.h>

// ---------------- problem constants ----------------
static constexpr int NN   = 4;
static constexpr int CC   = 256;
static constexpr int HH   = 64;
static constexpr int WW   = 64;
static constexpr int HWSZ = HH * WW;          // 4096
static constexpr int MTOT = NN * HWSZ;        // 16384 rows (NHWC)
static constexpr int GG   = 4;
static constexpr int GCC  = CC / GG;          // 64
static constexpr int PP   = 9;
static constexpr int PADP = 6;
static constexpr int DILD = 6;
static constexpr int HP   = HH + 2 * PADP;    // 76
static constexpr int WP   = WW + 2 * PADP;    // 76
static constexpr int XPADN = NN * HP * WP * CC;   // 5,914,624 floats
static constexpr float SLOPE = 0.01f;

// ---------------- vector types for WMMA ----------------
typedef __attribute__((ext_vector_type(16))) __bf16          v16bf;
typedef __attribute__((ext_vector_type(8)))  unsigned short  u16x8;
typedef __attribute__((ext_vector_type(8)))  float           v8f;

// ---------------- optional gfx1250 async-to-LDS path ----------------
#if defined(__HIP_DEVICE_COMPILE__) && __has_builtin(__builtin_amdgcn_global_load_async_to_lds_b128) && __has_builtin(__builtin_amdgcn_s_wait_asynccnt)
#define ASYNC_COPY 1
#else
#define ASYNC_COPY 0
#endif

#if ASYNC_COPY
typedef int v4i_g __attribute__((vector_size(4 * sizeof(int))));
typedef __attribute__((address_space(1))) v4i_g* gv4i_p;   // global
typedef __attribute__((address_space(3))) v4i_g* lv4i_p;   // LDS
#endif

__device__ __forceinline__ unsigned short f2bf(float f) {
  unsigned u = __float_as_uint(f);
  u += 0x7FFFu + ((u >> 16) & 1u);            // round-to-nearest-even
  return (unsigned short)(u >> 16);
}
__device__ __forceinline__ float bf2f(unsigned short h) {
  return __uint_as_float(((unsigned)h) << 16);
}

// Copy one 16B chunk of B into LDS (async if available)
__device__ __forceinline__ void cp_b128(const unsigned short* g, unsigned short* l) {
#if ASYNC_COPY
  __builtin_amdgcn_global_load_async_to_lds_b128((gv4i_p)(g), (lv4i_p)(l), 0, 0);
#else
  *(u16x8*)l = *(const u16x8*)g;
#endif
}
__device__ __forceinline__ void wait_async_le1() {
#if ASYNC_COPY
  __builtin_amdgcn_s_wait_asynccnt(1);
#endif
}
__device__ __forceinline__ void wait_async_le0() {
#if ASYNC_COPY
  __builtin_amdgcn_s_wait_asynccnt(0);
#endif
}

// Load a 16-element bf16 fragment: two contiguous 16B chunks (k..k+7, k+16..k+23)
__device__ __forceinline__ v16bf frag_load(const unsigned short* p) {
  union { u16x8 h[2]; v16bf v; } u;
  u.h[0] = *(const u16x8*)(p);
  u.h[1] = *(const u16x8*)(p + 16);
  return u.v;
}

// ---------------- prep: weight bf16 convert, BN fold, row map, SE accum zero ----------------
__global__ __launch_bounds__(256) void k_prep(
    const float* __restrict__ conv_w, const float* __restrict__ inproj_w,
    const float* __restrict__ outproj_w, const float* __restrict__ off_w,
    const float* __restrict__ mask_w, const float* __restrict__ fc1_w,
    const float* __restrict__ fc2_w,
    const float* __restrict__ bn1_g, const float* __restrict__ bn1_b,
    const float* __restrict__ bn1_m, const float* __restrict__ bn1_v,
    const float* __restrict__ dwbn_g, const float* __restrict__ dwbn_b,
    const float* __restrict__ dwbn_m, const float* __restrict__ dwbn_v,
    const float* __restrict__ bn2_g, const float* __restrict__ bn2_b,
    const float* __restrict__ bn2_m, const float* __restrict__ bn2_v,
    unsigned short* __restrict__ convw16, unsigned short* __restrict__ inprojw16,
    unsigned short* __restrict__ outprojw16, unsigned short* __restrict__ offw16,
    unsigned short* __restrict__ maskw16, unsigned short* __restrict__ fc1w16,
    unsigned short* __restrict__ fc2w16,
    float* __restrict__ bn1sc, float* __restrict__ bn1sh,
    float* __restrict__ dwsc,  float* __restrict__ dwsh,
    float* __restrict__ bn2sc, float* __restrict__ bn2sh,
    int* __restrict__ rowmap, float* __restrict__ svec)
{
  int i = blockIdx.x * 256 + threadIdx.x;
  if (i < 256 * 256) {
    convw16[i]   = f2bf(conv_w[i]);
    inprojw16[i] = f2bf(inproj_w[i]);
    outprojw16[i]= f2bf(outproj_w[i]);
  }
  if (i < 1024 * 256) {
    fc1w16[i] = f2bf(fc1_w[i]);
    fc2w16[i] = f2bf(fc2_w[i]);
  }
  if (i < 72 * 256) offw16[i]  = f2bf(off_w[i]);
  if (i < 36 * 256) maskw16[i] = f2bf(mask_w[i]);
  if (i < 256) {
    float s1 = bn1_g[i] * rsqrtf(bn1_v[i] + 1e-5f);
    bn1sc[i] = s1; bn1sh[i] = bn1_b[i] - bn1_m[i] * s1;
    float s2 = dwbn_g[i] * rsqrtf(dwbn_v[i] + 1e-5f);
    dwsc[i] = s2;  dwsh[i] = dwbn_b[i] - dwbn_m[i] * s2;
    float s3 = bn2_g[i] * rsqrtf(bn2_v[i] + 1e-5f);
    bn2sc[i] = s3; bn2sh[i] = bn2_b[i] - bn2_m[i] * s3;
  }
  if (i < MTOT) {
    int n = i >> 12, hw = i & 4095;
    int h = hw >> 6, w = hw & 63;
    rowmap[i] = (n * HP + h + PADP) * WP + (w + PADP);
  }
  if (i < NN * CC) svec[i] = 0.0f;
}

// ---------------- zero fill (xpad borders) ----------------
__global__ __launch_bounds__(256) void k_zero(float* __restrict__ p, int n) {
  int i = blockIdx.x * 256 + threadIdx.x;
  if (i < n) p[i] = 0.0f;
}

// ---------------- NCHW f32 -> NHWC bf16 (LDS tiled transpose) ----------------
__global__ __launch_bounds__(256) void k_nchw2nhwc(const float* __restrict__ x,
                                                   unsigned short* __restrict__ xT) {
  __shared__ float tile[32][33];
  int n = blockIdx.z, hw0 = blockIdx.x * 32, c0 = blockIdx.y * 32;
  #pragma unroll
  for (int i = 0; i < 32; i += 8) {
    int c = c0 + threadIdx.y + i;
    tile[threadIdx.y + i][threadIdx.x] =
        x[(size_t)(n * CC + c) * HWSZ + hw0 + threadIdx.x];
  }
  __syncthreads();
  #pragma unroll
  for (int i = 0; i < 32; i += 8) {
    int hw = hw0 + threadIdx.y + i;
    xT[(size_t)(n * HWSZ + hw) * CC + c0 + threadIdx.x] =
        f2bf(tile[threadIdx.x][threadIdx.y + i]);
  }
}

// ---------------- NHWC f32 -> NCHW f32 (final output) ----------------
__global__ __launch_bounds__(256) void k_nhwc2nchw(const float* __restrict__ nhwc,
                                                   float* __restrict__ out) {
  __shared__ float tile[32][33];
  int n = blockIdx.z, hw0 = blockIdx.x * 32, c0 = blockIdx.y * 32;
  #pragma unroll
  for (int i = 0; i < 32; i += 8) {
    int hw = hw0 + threadIdx.y + i;
    tile[threadIdx.y + i][threadIdx.x] =
        nhwc[(size_t)(n * HWSZ + hw) * CC + c0 + threadIdx.x];
  }
  __syncthreads();
  #pragma unroll
  for (int i = 0; i < 32; i += 8) {
    int c = c0 + threadIdx.y + i;
    out[(size_t)(n * CC + c) * HWSZ + hw0 + threadIdx.x] =
        tile[threadIdx.x][threadIdx.y + i];
  }
}

// ---------------- WMMA GEMM: C[M,Nout] = epilogue(A[M,K] * W[Nout,K]^T) ----------------
// Block: 256 threads = 8 waves; tile 128 rows x 64 cols.
// B tile (64x32 bf16, 4KB) staged in LDS (async DMA when available),
// double-buffered, shared by all 8 waves.
// MODE 0: BN(scale/shift)+leaky -> outF(f32) and outB(bf16)         (conv1)
// MODE 1: +bias, rows remapped via rowmap -> outF (padded buffer)   (inproj)
// MODE 2: +bias -> outF                                             (offset / mask logits)
// MODE 3: +bias +resid, then BN2 -> outF                            (outproj + bn2)
// MODE 4: +bias, leaky -> outB(bf16)                                (fc1)
// MODE 5: +bias +resid -> outF                                      (fc2 + residual)
template <int MODE, bool FULL>
__global__ __launch_bounds__(256) void k_gemm(
    const unsigned short* __restrict__ A, const unsigned short* __restrict__ Wm,
    const float* __restrict__ bias, const float* __restrict__ scale,
    const float* __restrict__ shift, const float* __restrict__ resid,
    const int* __restrict__ rowmap,
    float* __restrict__ outF, unsigned short* __restrict__ outB,
    int M, int K, int Nout)
{
  __shared__ unsigned short Bs[2][64 * 32];   // 2 x 4KB double buffer
  const int t    = threadIdx.x;
  const int lane = t & 31;
  const int wv   = t >> 5;
  const int rowBase = blockIdx.x * 128 + wv * 16;
  const int colBase = blockIdx.y * 64;
  const int r     = lane & 15;
  const int half8 = (lane >> 4) * 8;

  // cooperative B staging: each thread copies one 16B chunk per k-step
  const int brow = t >> 2;             // 0..63 : column within block col-tile
  const int bofs = (t & 3) * 8;        // element offset within 32-wide k slice
  const bool bvalid = FULL || (colBase + brow < Nout);
  const unsigned short* gB = Wm + (size_t)(colBase + brow) * K + bofs;
  unsigned short* lB0 = &Bs[0][brow * 32 + bofs];
  unsigned short* lB1 = &Bs[1][brow * 32 + bofs];

  const unsigned short* pa = A + (size_t)(rowBase + r) * K + half8;

  const int nsteps = K >> 5;
  v8f acc0 = (v8f)0.0f, acc1 = (v8f)0.0f, acc2 = (v8f)0.0f, acc3 = (v8f)0.0f;

  if (bvalid) cp_b128(gB, lB0);
  for (int s = 0; s < nsteps; ++s) {
    if (s + 1 < nsteps) {
      if (bvalid) cp_b128(gB + (size_t)(s + 1) * 32, (s & 1) ? lB0 : lB1);
      wait_async_le1();
    } else {
      wait_async_le0();
    }
    __syncthreads();

    v16bf a = frag_load(pa + s * 32);
    const unsigned short* bs = Bs[s & 1];
    v16bf b0 = frag_load(bs + (r +  0) * 32 + half8);
    v16bf b1 = frag_load(bs + (r + 16) * 32 + half8);
    v16bf b2 = frag_load(bs + (r + 32) * 32 + half8);
    v16bf b3 = frag_load(bs + (r + 48) * 32 + half8);
    acc0 = __builtin_amdgcn_wmma_f32_16x16x32_bf16(false, a, false, b0, (short)0, acc0, false, false);
    acc1 = __builtin_amdgcn_wmma_f32_16x16x32_bf16(false, a, false, b1, (short)0, acc1, false, false);
    acc2 = __builtin_amdgcn_wmma_f32_16x16x32_bf16(false, a, false, b2, (short)0, acc2, false, false);
    acc3 = __builtin_amdgcn_wmma_f32_16x16x32_bf16(false, a, false, b3, (short)0, acc3, false, false);

    if (s + 1 < nsteps) __syncthreads();   // protect next overwrite of consumed buffer
  }

  const int hiAdd = (lane >> 4) * 8;
  auto epi = [&](v8f acc, int ct) {
    int gc = colBase + ct * 16 + r;
    if (!FULL && gc >= Nout) return;
    #pragma unroll
    for (int v = 0; v < 8; v++) {
      int gr = rowBase + v + hiAdd;
      float val = acc[v];
      if constexpr (MODE == 0) {
        val = val * scale[gc] + shift[gc];
        val = val > 0.0f ? val : val * SLOPE;
        outF[(size_t)gr * Nout + gc] = val;
        outB[(size_t)gr * Nout + gc] = f2bf(val);
      } else if constexpr (MODE == 1) {
        val += bias[gc];
        outF[(size_t)rowmap[gr] * CC + gc] = val;
      } else if constexpr (MODE == 2) {
        val += bias[gc];
        outF[(size_t)gr * Nout + gc] = val;
      } else if constexpr (MODE == 3) {
        val += bias[gc] + resid[(size_t)gr * Nout + gc];
        val = val * scale[gc] + shift[gc];
        outF[(size_t)gr * Nout + gc] = val;
      } else if constexpr (MODE == 4) {
        val += bias[gc];
        val = val > 0.0f ? val : val * SLOPE;
        outB[(size_t)gr * Nout + gc] = f2bf(val);
      } else {
        val += bias[gc] + resid[(size_t)gr * CC + gc];
        outF[(size_t)gr * Nout + gc] = val;
      }
    }
  };
  epi(acc0, 0); epi(acc1, 1); epi(acc2, 2); epi(acc3, 3);
}

// ---------------- depthwise 3x3 + BN + leaky (NHWC bf16 in/out) ----------------
__global__ __launch_bounds__(256) void k_dwconv(
    const unsigned short* __restrict__ y, const float* __restrict__ dw_w,
    const float* __restrict__ dw_b, const float* __restrict__ sc,
    const float* __restrict__ sh, unsigned short* __restrict__ t)
{
  int idx = blockIdx.x * 256 + threadIdx.x;   // idx = m*256 + c
  if (idx >= MTOT * CC) return;
  int c = idx & 255;
  int m = idx >> 8;
  int n = m >> 12, hw = m & 4095;
  int h = hw >> 6, w = hw & 63;
  float acc = dw_b[c];
  #pragma unroll
  for (int i = 0; i < 3; i++) {
    #pragma unroll
    for (int j = 0; j < 3; j++) {
      int hh = h + i - 1, ww = w + j - 1;
      if (hh >= 0 && hh < HH && ww >= 0 && ww < WW) {
        acc += bf2f(y[(size_t)((n * HWSZ) + hh * WW + ww) * CC + c]) *
               dw_w[c * 9 + i * 3 + j];
      }
    }
  }
  float v = acc * sc[c] + sh[c];
  v = v > 0.0f ? v : v * SLOPE;
  t[idx] = f2bf(v);
}

// ---------------- softmax over P=9 (in-place on [M, G*P]) ----------------
__global__ __launch_bounds__(256) void k_softmax9(float* __restrict__ m) {
  int idx = blockIdx.x * 256 + threadIdx.x;   // M*G
  if (idx >= MTOT * GG) return;
  float* p = m + (size_t)(idx >> 2) * (GG * PP) + (idx & 3) * PP;
  float mx = p[0];
  #pragma unroll
  for (int i = 1; i < 9; i++) mx = fmaxf(mx, p[i]);
  float s = 0.0f, e[9];
  #pragma unroll
  for (int i = 0; i < 9; i++) { e[i] = expf(p[i] - mx); s += e[i]; }
  float inv = 1.0f / s;
  #pragma unroll
  for (int i = 0; i < 9; i++) p[i] = e[i] * inv;
}

// ---------------- deformable bilinear sampling ----------------
__global__ __launch_bounds__(256) void k_sample(
    const float* __restrict__ xpad, const float* __restrict__ offset,
    const float* __restrict__ mask, unsigned short* __restrict__ dout)
{
  int idx = blockIdx.x * 256 + threadIdx.x;   // ((m*G)+g)*GC + c
  if (idx >= MTOT * CC) return;
  int c = idx & 63;
  int g = (idx >> 6) & 3;
  int m = idx >> 8;
  int n = m >> 12, hw = m & 4095;
  int h = hw >> 6, w = hw & 63;
  const float* off = offset + (size_t)m * (GG * PP * 2) + g * (PP * 2);
  const float* mk  = mask   + (size_t)m * (GG * PP)     + g * PP;
  const float* img = xpad + (size_t)n * HP * WP * CC + g * GCC + c;
  float acc = 0.0f;
  #pragma unroll
  for (int p = 0; p < 9; p++) {
    int di = (p / 3) - 1, dj = (p % 3) - 1;   // torch x-major: x uses d[i], y uses d[j]
    float gx = (float)(w + PADP + di * DILD) + off[p * 2 + 0];
    float gy = (float)(h + PADP + dj * DILD) + off[p * 2 + 1];
    float fx = floorf(gx), fy = floorf(gy);
    int x0 = (int)fx, y0 = (int)fy;
    float wx = gx - fx, wy = gy - fy;
    float v00 = 0.f, v10 = 0.f, v01 = 0.f, v11 = 0.f;
    bool xa = (x0 >= 0 && x0 < WP), xb = (x0 + 1 >= 0 && x0 + 1 < WP);
    bool ya = (y0 >= 0 && y0 < HP), yb = (y0 + 1 >= 0 && y0 + 1 < HP);
    if (xa && ya) v00 = img[(size_t)(y0 * WP + x0) * CC];
    if (xb && ya) v10 = img[(size_t)(y0 * WP + x0 + 1) * CC];
    if (xa && yb) v01 = img[(size_t)((y0 + 1) * WP + x0) * CC];
    if (xb && yb) v11 = img[(size_t)((y0 + 1) * WP + x0 + 1) * CC];
    float bil = v00 * (1.f - wx) * (1.f - wy) + v10 * wx * (1.f - wy) +
                v01 * (1.f - wx) * wy + v11 * wx * wy;
    acc += mk[p] * bil;
  }
  dout[(size_t)m * CC + g * GCC + c] = f2bf(acc);
}

// ---------------- SE: partial channel sums (atomic) ----------------
__global__ __launch_bounds__(256) void k_se_reduce(const float* __restrict__ x2,
                                                   float* __restrict__ s) {
  int n = blockIdx.x >> 5;
  int chunk = blockIdx.x & 31;
  int c = threadIdx.x;
  float acc = 0.0f;
  for (int rr = 0; rr < 128; rr++) {
    int hw = chunk * 128 + rr;
    acc += x2[(size_t)(n * HWSZ + hw) * CC + c];
  }
  atomicAdd(&s[n * CC + c], acc);
}

// ---------------- SE: gate = sigmoid(relu(mean @ dw^T + db) @ uw^T + ub) ----------------
__global__ __launch_bounds__(256) void k_se_gate(
    const float* __restrict__ s, const float* __restrict__ dw,
    const float* __restrict__ db, const float* __restrict__ uw,
    const float* __restrict__ ub, float* __restrict__ gate)
{
  __shared__ float hid[NN * 64];
  int t = threadIdx.x;
  int n = t >> 6, o = t & 63;
  float a = db[o];
  for (int k = 0; k < CC; k++)
    a += (s[n * CC + k] * (1.0f / (float)HWSZ)) * dw[o * CC + k];
  hid[t] = fmaxf(a, 0.0f);
  __syncthreads();
  for (int n2 = 0; n2 < NN; n2++) {
    float a2 = ub[t];
    for (int k = 0; k < 64; k++) a2 += hid[n2 * 64 + k] * uw[t * 64 + k];
    gate[n2 * CC + t] = 1.0f / (1.0f + expf(-a2));
  }
}

// ---------------- x3 = x2 * gate -> bf16 ----------------
__global__ __launch_bounds__(256) void k_gate_mul(const float* __restrict__ x2,
                                                  const float* __restrict__ gate,
                                                  unsigned short* __restrict__ x3) {
  int idx = blockIdx.x * 256 + threadIdx.x;
  if (idx >= MTOT * CC) return;
  int c = idx & 255;
  int n = idx >> 20;
  x3[idx] = f2bf(x2[idx] * gate[n * CC + c]);
}

// =======================================================================
extern "C" void kernel_launch(void* const* d_in, const int* in_sizes, int n_in,
                              void* d_out, int out_size, void* d_ws, size_t ws_size,
                              hipStream_t stream) {
  (void)in_sizes; (void)n_in; (void)out_size; (void)ws_size;

  const float* x        = (const float*)d_in[0];
  const float* conv_w   = (const float*)d_in[1];
  const float* bn1_g    = (const float*)d_in[2];
  const float* bn1_b    = (const float*)d_in[3];
  const float* bn1_m    = (const float*)d_in[4];
  const float* bn1_v    = (const float*)d_in[5];
  const float* dw_w     = (const float*)d_in[6];
  const float* dw_b     = (const float*)d_in[7];
  const float* dwbn_g   = (const float*)d_in[8];
  const float* dwbn_b   = (const float*)d_in[9];
  const float* dwbn_m   = (const float*)d_in[10];
  const float* dwbn_v   = (const float*)d_in[11];
  const float* off_w    = (const float*)d_in[12];
  const float* off_b    = (const float*)d_in[13];
  const float* mask_w   = (const float*)d_in[14];
  const float* mask_b   = (const float*)d_in[15];
  const float* inproj_w = (const float*)d_in[16];
  const float* inproj_b = (const float*)d_in[17];
  const float* outproj_w= (const float*)d_in[18];
  const float* outproj_b= (const float*)d_in[19];
  const float* bn2_g    = (const float*)d_in[20];
  const float* bn2_b    = (const float*)d_in[21];
  const float* bn2_m    = (const float*)d_in[22];
  const float* bn2_v    = (const float*)d_in[23];
  const float* se_dw    = (const float*)d_in[24];
  const float* se_db    = (const float*)d_in[25];
  const float* se_uw    = (const float*)d_in[26];
  const float* se_ub    = (const float*)d_in[27];
  const float* fc1_w    = (const float*)d_in[28];
  const float* fc1_b    = (const float*)d_in[29];
  const float* fc2_w    = (const float*)d_in[30];
  const float* fc2_b    = (const float*)d_in[31];
  float* out = (float*)d_out;

  // ---- workspace layout ----
  char* ws = (char*)d_ws;
  size_t cur = 0;
  auto alloc = [&](size_t bytes) -> void* {
    void* p = ws + cur;
    cur += (bytes + 255) & ~(size_t)255;
    return p;
  };
  unsigned short* convw16   = (unsigned short*)alloc(256 * 256 * 2);
  unsigned short* inprojw16 = (unsigned short*)alloc(256 * 256 * 2);
  unsigned short* outprojw16= (unsigned short*)alloc(256 * 256 * 2);
  unsigned short* offw16    = (unsigned short*)alloc(72 * 256 * 2);
  unsigned short* maskw16   = (unsigned short*)alloc(36 * 256 * 2);
  unsigned short* fc1w16    = (unsigned short*)alloc(1024 * 256 * 2);
  unsigned short* fc2w16    = (unsigned short*)alloc(256 * 1024 * 2);
  float* bn1sc = (float*)alloc(256 * 4);
  float* bn1sh = (float*)alloc(256 * 4);
  float* dwsc  = (float*)alloc(256 * 4);
  float* dwsh  = (float*)alloc(256 * 4);
  float* bn2sc = (float*)alloc(256 * 4);
  float* bn2sh = (float*)alloc(256 * 4);
  int*   rowmap = (int*)alloc(MTOT * 4);
  unsigned short* xT   = (unsigned short*)alloc((size_t)MTOT * CC * 2); // also t_bf16
  float* inp           = (float*)alloc((size_t)MTOT * CC * 4);
  unsigned short* ybf  = (unsigned short*)alloc((size_t)MTOT * CC * 2); // also x3_bf16
  float* xpad          = (float*)alloc((size_t)XPADN * 4);
  float* offset        = (float*)alloc((size_t)MTOT * 72 * 4);
  float* maskb         = (float*)alloc((size_t)MTOT * 36 * 4);
  unsigned short* dout = (unsigned short*)alloc((size_t)MTOT * CC * 2);
  float* x2            = (float*)alloc((size_t)MTOT * CC * 4);          // also out_nhwc
  float* svec          = (float*)alloc(NN * CC * 4);
  float* gate          = (float*)alloc(NN * CC * 4);
  unsigned short* hbf  = (unsigned short*)alloc((size_t)MTOT * 1024 * 2);
  unsigned short* tbf  = xT;   // alias: xT dead after conv1 GEMM
  unsigned short* x3bf = ybf;  // alias: y dead after dwconv
  float* out_nhwc      = x2;   // alias: x2 dead after gate_mul

  // ---- 1. prep ----
  k_prep<<<(1024 * 256 + 255) / 256, 256, 0, stream>>>(
      conv_w, inproj_w, outproj_w, off_w, mask_w, fc1_w, fc2_w,
      bn1_g, bn1_b, bn1_m, bn1_v, dwbn_g, dwbn_b, dwbn_m, dwbn_v,
      bn2_g, bn2_b, bn2_m, bn2_v,
      convw16, inprojw16, outprojw16, offw16, maskw16, fc1w16, fc2w16,
      bn1sc, bn1sh, dwsc, dwsh, bn2sc, bn2sh, rowmap, svec);

  // ---- 2. zero xpad (borders must be 0 each call) ----
  k_zero<<<(XPADN + 255) / 256, 256, 0, stream>>>(xpad, XPADN);

  // ---- 3. x NCHW -> NHWC bf16 ----
  k_nchw2nhwc<<<dim3(HWSZ / 32, CC / 32, NN), dim3(32, 8), 0, stream>>>(x, xT);

  // ---- 4. conv1 + BN1 + leaky -> inp(f32), y(bf16) ----
  k_gemm<0, true><<<dim3(MTOT / 128, CC / 64), 256, 0, stream>>>(
      xT, convw16, nullptr, bn1sc, bn1sh, nullptr, nullptr, inp, ybf,
      MTOT, CC, CC);

  // ---- 5. inproj -> xpad interior ----
  k_gemm<1, true><<<dim3(MTOT / 128, CC / 64), 256, 0, stream>>>(
      ybf, inprojw16, inproj_b, nullptr, nullptr, nullptr, rowmap, xpad, nullptr,
      MTOT, CC, CC);

  // ---- 6. depthwise 3x3 + dwbn + leaky -> t(bf16) ----
  k_dwconv<<<(MTOT * CC + 255) / 256, 256, 0, stream>>>(
      ybf, dw_w, dw_b, dwsc, dwsh, tbf);

  // ---- 7. offset head (Nout=72) ----
  k_gemm<2, false><<<dim3(MTOT / 128, 2), 256, 0, stream>>>(
      tbf, offw16, off_b, nullptr, nullptr, nullptr, nullptr, offset, nullptr,
      MTOT, CC, 72);

  // ---- 8. mask head (Nout=36) + softmax over 9 ----
  k_gemm<2, false><<<dim3(MTOT / 128, 1), 256, 0, stream>>>(
      tbf, maskw16, mask_b, nullptr, nullptr, nullptr, nullptr, maskb, nullptr,
      MTOT, CC, 36);
  k_softmax9<<<(MTOT * GG + 255) / 256, 256, 0, stream>>>(maskb);

  // ---- 9. deformable sampling -> dout(bf16) ----
  k_sample<<<(MTOT * CC + 255) / 256, 256, 0, stream>>>(xpad, offset, maskb, dout);

  // ---- 10. outproj + residual(inp) + BN2 -> x2(f32) ----
  k_gemm<3, true><<<dim3(MTOT / 128, CC / 64), 256, 0, stream>>>(
      dout, outprojw16, outproj_b, bn2sc, bn2sh, inp, nullptr, x2, nullptr,
      MTOT, CC, CC);

  // ---- 11. SE ----
  k_se_reduce<<<NN * 32, 256, 0, stream>>>(x2, svec);
  k_se_gate<<<1, 256, 0, stream>>>(svec, se_dw, se_db, se_uw, se_ub, gate);
  k_gate_mul<<<(MTOT * CC + 255) / 256, 256, 0, stream>>>(x2, gate, x3bf);

  // ---- 12. fc1 (256->1024) + leaky -> h(bf16) ----
  k_gemm<4, true><<<dim3(MTOT / 128, 1024 / 64), 256, 0, stream>>>(
      x3bf, fc1w16, fc1_b, nullptr, nullptr, nullptr, nullptr, nullptr, hbf,
      MTOT, CC, 1024);

  // ---- 13. fc2 (1024->256) + residual(inp) -> out_nhwc(f32) ----
  k_gemm<5, true><<<dim3(MTOT / 128, CC / 64), 256, 0, stream>>>(
      hbf, fc2w16, fc2_b, nullptr, nullptr, inp, nullptr, out_nhwc, nullptr,
      MTOT, 1024, CC);

  // ---- 14. NHWC -> NCHW final ----
  k_nhwc2nchw<<<dim3(HWSZ / 32, CC / 32, NN), dim3(32, 8), 0, stream>>>(out_nhwc, out);
}
